// CWRRTWindowCell_6828998001264
// MI455X (gfx1250) — matrix operand order repaired and checked by reading, not compiled, not run on hardware
//
#include <hip/hip_runtime.h>
#include <hip/hip_bf16.h>
#include <math.h>

#define DEVFN __device__ __forceinline__

typedef __attribute__((ext_vector_type(16))) __bf16 v16bf;
typedef __attribute__((ext_vector_type(8)))  float  v8f;
typedef unsigned int uint4v __attribute__((ext_vector_type(4)));
typedef int          int4v  __attribute__((ext_vector_type(4)));
typedef int          int8v  __attribute__((ext_vector_type(8)));

constexpr int B  = 4;
constexpr int T  = 1024;
constexpr int O  = 128;
constexpr int C  = 512;
constexpr int H  = 8;
constexpr int L  = 6;
constexpr int V  = 32000;
constexpr int HD = 64;       // C / H
constexpr int KV = T + O;    // 1152

DEVFN __bf16 f2bf(float f) {
  union { float f; unsigned u; } a; a.f = f;
  unsigned r = a.u + 0x7FFFu + ((a.u >> 16) & 1u);
  union { unsigned short s; __bf16 b; } o; o.s = (unsigned short)(r >> 16);
  return o.b;
}

DEVFN v8f zero8() { v8f z = {0.f,0.f,0.f,0.f,0.f,0.f,0.f,0.f}; return z; }

// 16-bit A-matrix 16x32 fragment: VGPR-pair index v, lane half -> K base
DEVFN int a_k_of(int v, int half) { return ((v >> 2) << 4) + (half << 3) + ((v & 3) << 1); }
// 16-bit B-matrix 32x16 fragment: K base for VGPR-pair v, lane half
DEVFN int b_k_of(int v, int half) { return ((v >> 2) << 4) + (half << 3) + ((v & 3) << 1); }

DEVFN void wait_tensorcnt0() {
#if __has_builtin(__builtin_amdgcn_s_wait_tensorcnt)
  __builtin_amdgcn_s_wait_tensorcnt(0);
#else
  asm volatile("s_wait_tensorcnt 0" ::: "memory");
#endif
}

// ---------------------------------------------------------------------------
// Generic bf16-WMMA GEMM: Out[M,N] = act(A[M,K] @ Bm[K,N] + bias) + res
// Block: 256 threads = 8 waves in a 2(M)x4(N) grid; block tile 64(M) x 128(N);
// each wave owns a 32x32 output (2x2 WMMA tiles -> 4 WMMAs per K-step).
// B tile (32x128 fp32) is DMA'd into LDS by the Tensor Data Mover when
// available; A tile staged manually as bf16. Requires N%128==0, K%32==0.
// ---------------------------------------------------------------------------
__global__ __launch_bounds__(256)
void gemm_wmma(const float* __restrict__ A, const float* __restrict__ Bm,
               const float* __restrict__ bias, const float* __restrict__ res,
               float* __restrict__ Out, int M, int N, int K, int act) {
  __shared__ __bf16 lA[64 * 34];    // [m][k] bf16, stride 34
  __shared__ float  lBf[32 * 128];  // [k][n] fp32 (TDM destination)
  const int tid  = threadIdx.x;
  const int lane = tid & 31;
  const int w    = tid >> 5;
  const int half = lane >> 4;
  const int q    = lane & 15;
  const int wm   = w & 1;           // 0..1 -> 32-row M sub-tile
  const int wn   = w >> 1;          // 0..3 -> 32-col N sub-tile
  const int m0   = blockIdx.y * 64;
  const int n0   = blockIdx.x * 128;
  const bool fullM = (m0 + 64 <= M);

  v8f acc[2][2];
#pragma unroll
  for (int mi = 0; mi < 2; ++mi)
#pragma unroll
    for (int ni = 0; ni < 2; ++ni) acc[mi][ni] = zero8();

  for (int k0 = 0; k0 < K; k0 += 32) {
    // ---- stage B tile 32x128 fp32 into LDS ----
#if __has_builtin(__builtin_amdgcn_tensor_load_to_lds)
    if (w == 0) {
      // Tensor DMA descriptor (D#), 2D tensor [K x N] fp32, tile 128x32.
      unsigned long long ga =
          (unsigned long long)(uintptr_t)(Bm + (size_t)k0 * N + n0);
      unsigned ldsa = (unsigned)(uintptr_t)(&lBf[0]);   // LDS aperture: low 32 = offset
      uint4v g0;
      g0[0] = 1u;                                        // count=1 valid descriptor
      g0[1] = ldsa;                                      // lds_addr
      g0[2] = (unsigned)ga;                              // global_addr[31:0]
      g0[3] = (unsigned)((ga >> 32) & 0x01FFFFFFu) | 0x80000000u; // addr[56:32] | type=2
      unsigned td0 = (unsigned)N;                        // tensor_dim0 (elements)
      unsigned td1 = (unsigned)K;                        // tensor_dim1 (rows)
      int8v g1;
      g1[0] = (int)0x00020000u;                          // data_size=2 (4 bytes)
      g1[1] = (int)((td0 & 0xFFFFu) << 16);              // tensor_dim0[15:0]
      g1[2] = (int)((td0 >> 16) | ((td1 & 0xFFFFu) << 16));
      g1[3] = (int)((td1 >> 16) | (128u << 16));         // tile_dim0 = 128
      g1[4] = (int)32u;                                  // tile_dim1 = 32
      g1[5] = (int)td0;                                  // tensor_dim0_stride = N
      g1[6] = 0; g1[7] = 0;
      int4v gz; gz[0] = 0; gz[1] = 0; gz[2] = 0; gz[3] = 0;
#if __clang_major__ >= 23
      int8v gz8; gz8[0]=0; gz8[1]=0; gz8[2]=0; gz8[3]=0; gz8[4]=0; gz8[5]=0; gz8[6]=0; gz8[7]=0;
      __builtin_amdgcn_tensor_load_to_lds(g0, g1, gz, gz, gz8, 0);
#else
      __builtin_amdgcn_tensor_load_to_lds(g0, g1, gz, gz, 0);
#endif
    }
#else
#pragma unroll
    for (int i = 0; i < 16; ++i) {
      int e = tid + i * 256;
      int kk = e >> 7, n = e & 127;
      lBf[e] = Bm[(size_t)(k0 + kk) * N + n0 + n];
    }
#endif
    // ---- stage A tile 64x32 bf16 (uniform fast path when fully interior) ----
    if (fullM) {
#pragma unroll
      for (int i = 0; i < 8; ++i) {
        int e = tid + i * 256;
        int m = e >> 5, kk = e & 31;
        lA[m * 34 + kk] = f2bf(A[(size_t)(m0 + m) * K + k0 + kk]);
      }
    } else {
#pragma unroll
      for (int i = 0; i < 8; ++i) {
        int e = tid + i * 256;
        int m = e >> 5, kk = e & 31;
        int mg = m0 + m;
        float val = (mg < M) ? A[(size_t)mg * K + k0 + kk] : 0.f;
        lA[m * 34 + kk] = f2bf(val);
      }
    }
    if (k0 + 32 < K) {  // cache hint for next K tile (global_prefetch_b8)
      int mg = m0 + (tid >> 2);
      if (mg < M) __builtin_prefetch(&A[(size_t)mg * K + k0 + 32 + ((tid & 3) << 3)], 0, 1);
      __builtin_prefetch(&Bm[(size_t)(k0 + 32 + (tid >> 7)) * N + n0 + (tid & 127)], 0, 1);
    }
#if __has_builtin(__builtin_amdgcn_tensor_load_to_lds)
    if (w == 0) wait_tensorcnt0();   // TDM completion before anyone consumes lBf
#endif
    __syncthreads();

    // ---- fragments + 4 WMMAs ----
    v16bf af[2], bf[2];
#pragma unroll
    for (int mi = 0; mi < 2; ++mi) {
      const int r = wm * 32 + mi * 16 + q;     // A row (m = lane&15)
#pragma unroll
      for (int v = 0; v < 8; ++v) {
        int ak = a_k_of(v, half);
        const __bf16* ap = &lA[r * 34 + ak];
        af[mi][2 * v] = ap[0]; af[mi][2 * v + 1] = ap[1];
      }
    }
#pragma unroll
    for (int ni = 0; ni < 2; ++ni) {
      const int n = wn * 32 + ni * 16 + q;     // B col (n = lane&15)
#pragma unroll
      for (int v = 0; v < 8; ++v) {
        int kk = b_k_of(v, half);
        bf[ni][2 * v]     = f2bf(lBf[kk * 128 + n]);
        bf[ni][2 * v + 1] = f2bf(lBf[(kk + 1) * 128 + n]);
      }
    }
#pragma unroll
    for (int mi = 0; mi < 2; ++mi)
#pragma unroll
      for (int ni = 0; ni < 2; ++ni)
        acc[mi][ni] = __builtin_amdgcn_wmma_f32_16x16x32_bf16(
            false, af[mi], false, bf[ni], (short)0, acc[mi][ni], false, false);
    __syncthreads();
  }

  // ---- epilogue ----
#pragma unroll
  for (int ni = 0; ni < 2; ++ni) {
    const int ng = n0 + wn * 32 + ni * 16 + q;
    const float bv = bias ? bias[ng] : 0.f;
#pragma unroll
    for (int mi = 0; mi < 2; ++mi) {
#pragma unroll
      for (int v = 0; v < 8; ++v) {
        int mg = m0 + wm * 32 + mi * 16 + v + 8 * half;
        if (mg < M) {
          float r = acc[mi][ni][v] + bv;
          if (act) {  // tanh-approx GELU
            float x3 = r * r * r;
            r = 0.5f * r * (1.f + tanhf(0.7978845608028654f * (r + 0.044715f * x3)));
          }
          if (res) r += res[(size_t)mg * N + ng];
          Out[(size_t)mg * N + ng] = r;
        }
      }
    }
  }
}

// ---------------------------------------------------------------------------
// Flash-style attention. One wave per (b, h, 16-query tile).
// Q: [B,T,C] roped+post-proj; Kb/Vb: [B,KV,C]; Outp: [B,T,C].
// No __syncthreads (waves run different trip counts); per-wave LDS for P.
// ---------------------------------------------------------------------------
__global__ __launch_bounds__(256)
void attn_wmma(const float* __restrict__ Q, const float* __restrict__ Kb,
               const float* __restrict__ Vb, const int* __restrict__ tok,
               float* __restrict__ Outp) {
  __shared__ __bf16 pbuf[8][16 * 34];   // per-wave 16x32 P tile, stride 34
  const int tid  = threadIdx.x;
  const int lane = tid & 31;
  const int w    = tid >> 5;
  const int half = lane >> 4;
  const int q    = lane & 15;
  const int b    = blockIdx.z;
  const int h    = blockIdx.y;
  const int qt   = blockIdx.x * 8 + w;  // 0..63
  const int q0   = qt * 16;

  const float scale = 0.125f;           // 1/sqrt(HD)

  // Q fragments (A layout, row m = lane&15), scale folded in.
  const float* qrow = Q + ((size_t)(b * T + q0 + q)) * C + h * HD;
  v16bf a_lo, a_hi;
#pragma unroll
  for (int v = 0; v < 8; ++v) {
    int d = a_k_of(v, half);
    float2 p0 = *(const float2*)(qrow + d);
    float2 p1 = *(const float2*)(qrow + d + 32);
    a_lo[2 * v] = f2bf(p0.x * scale); a_lo[2 * v + 1] = f2bf(p0.y * scale);
    a_hi[2 * v] = f2bf(p1.x * scale); a_hi[2 * v + 1] = f2bf(p1.y * scale);
  }

  float rm[8], rs[8];
#pragma unroll
  for (int v = 0; v < 8; ++v) { rm[v] = -1e30f; rs[v] = 0.f; }
  v8f acc[4];
#pragma unroll
  for (int dt = 0; dt < 4; ++dt) acc[dt] = zero8();

  const int kmax    = O + q0 + 16;          // exclusive causal bound for this tile
  const int nchunks = (kmax + 31) >> 5;

  for (int ck = 0; ck < nchunks; ++ck) {
    const int kbase = ck << 5;
    v8f S[2];
#pragma unroll
    for (int sub = 0; sub < 2; ++sub) {
      const int kg = kbase + sub * 16 + q;  // this lane's key column
      const float* krow = Kb + ((size_t)(b * KV + kg)) * C + h * HD;
      v16bf b_lo, b_hi;
#pragma unroll
      for (int v = 0; v < 8; ++v) {
        int d = b_k_of(v, half);
        float2 p0 = *(const float2*)(krow + d);
        float2 p1 = *(const float2*)(krow + d + 32);
        b_lo[2 * v] = f2bf(p0.x); b_lo[2 * v + 1] = f2bf(p0.y);
        b_hi[2 * v] = f2bf(p1.x); b_hi[2 * v + 1] = f2bf(p1.y);
      }
      v8f s = zero8();
      s = __builtin_amdgcn_wmma_f32_16x16x32_bf16(false, a_lo, false, b_lo,
                                                  (short)0, s, false, false);
      s = __builtin_amdgcn_wmma_f32_16x16x32_bf16(false, a_hi, false, b_hi,
                                                  (short)0, s, false, false);
      // mask: memory keys always valid; token keys need causal + non-pad
      const bool kvalid = (kg < O) || (tok[b * T + (kg - O)] != 0);
#pragma unroll
      for (int v = 0; v < 8; ++v) {
        int qg = q0 + v + 8 * half;
        bool ok = kvalid && ((kg < O) || ((kg - O) <= qg));
        s[v] = ok ? s[v] : -1e9f;
      }
      S[sub] = s;
    }

    // online softmax per row (row stats replicated across 16 lanes of a half)
    float cf[8];
#pragma unroll
    for (int v = 0; v < 8; ++v) {
      float mx = fmaxf(S[0][v], S[1][v]);
#pragma unroll
      for (int off = 1; off < 16; off <<= 1) mx = fmaxf(mx, __shfl_xor(mx, off, 32));
      float nm = fmaxf(rm[v], mx);
      float c  = expf(rm[v] - nm);
      float p0 = expf(S[0][v] - nm);
      float p1 = expf(S[1][v] - nm);
      float sm = p0 + p1;
#pragma unroll
      for (int off = 1; off < 16; off <<= 1) sm += __shfl_xor(sm, off, 32);
      rs[v] = rs[v] * c + sm;
      rm[v] = nm;
      cf[v] = c;
      int m = v + 8 * half;
      pbuf[w][m * 34 + q]      = f2bf(p0);
      pbuf[w][m * 34 + 16 + q] = f2bf(p1);
    }
#pragma unroll
    for (int dt = 0; dt < 4; ++dt) {
#pragma unroll
      for (int v = 0; v < 8; ++v) acc[dt][v] *= cf[v];
    }

    // DS writes must land before the cross-lane re-layout reads
    asm volatile("s_wait_dscnt 0" ::: "memory");

    v16bf pa;  // P as 16x32 A fragment
#pragma unroll
    for (int v = 0; v < 8; ++v) {
      int kk = a_k_of(v, half);
      const __bf16* pp = &pbuf[w][q * 34 + kk];
      pa[2 * v] = pp[0]; pa[2 * v + 1] = pp[1];
    }
#pragma unroll
    for (int dt = 0; dt < 4; ++dt) {
      v16bf vf;
#pragma unroll
      for (int v = 0; v < 8; ++v) {
        int kk = b_k_of(v, half);
        const float* vp = Vb + ((size_t)(b * KV + kbase + kk)) * C + h * HD + dt * 16 + q;
        vf[2 * v]     = f2bf(vp[0]);
        vf[2 * v + 1] = f2bf(vp[C]);
      }
      acc[dt] = __builtin_amdgcn_wmma_f32_16x16x32_bf16(false, pa, false, vf,
                                                        (short)0, acc[dt], false, false);
    }
  }

#pragma unroll
  for (int v = 0; v < 8; ++v) {
    int m = v + 8 * half;
    float inv = 1.f / rs[v];
    float* op = Outp + ((size_t)(b * T + q0 + m)) * C + h * HD + q;
#pragma unroll
    for (int dt = 0; dt < 4; ++dt) op[dt * 16] = acc[dt][v] * inv;
  }
}

// ---------------------------------------------------------------------------
// Elementwise / reduction kernels
// ---------------------------------------------------------------------------
__global__ __launch_bounds__(256)
void ln_kernel(const float* __restrict__ in, const float* __restrict__ sc,
               const float* __restrict__ bi, float* __restrict__ out) {
  __shared__ float red[256];
  const int row = blockIdx.x;
  const float* x = in + (size_t)row * C;
  float s = 0.f;
  for (int c = threadIdx.x; c < C; c += 256) s += x[c];
  red[threadIdx.x] = s; __syncthreads();
  for (int st = 128; st > 0; st >>= 1) {
    if (threadIdx.x < st) red[threadIdx.x] += red[threadIdx.x + st];
    __syncthreads();
  }
  const float mean = red[0] / (float)C;
  __syncthreads();
  float vv = 0.f;
  for (int c = threadIdx.x; c < C; c += 256) { float d = x[c] - mean; vv += d * d; }
  red[threadIdx.x] = vv; __syncthreads();
  for (int st = 128; st > 0; st >>= 1) {
    if (threadIdx.x < st) red[threadIdx.x] += red[threadIdx.x + st];
    __syncthreads();
  }
  const float inv = rsqrtf(red[0] / (float)C + 1e-6f);
  float* o = out + (size_t)row * C;
  for (int c = threadIdx.x; c < C; c += 256)
    o[c] = (x[c] - mean) * inv * sc[c] + bi[c];
}

__global__ __launch_bounds__(256)
void embed_kernel(const int* __restrict__ tok, const float* __restrict__ emb,
                  const float* __restrict__ ssp, const float* __restrict__ alpha,
                  float* __restrict__ x) {
  size_t i = (size_t)blockIdx.x * 256 + threadIdx.x;   // B*T*C total
  int c = (int)(i % C);
  size_t bt = i / C;
  int b = (int)(bt / T);
  float a = 1.f / (1.f + expf(-alpha[c]));
  x[i] = emb[(size_t)tok[bt] * C + c] + ssp[b * C + c] * a;
}

__global__ __launch_bounds__(256)
void concat_kernel(const float* __restrict__ memp, const float* __restrict__ cur,
                   float* __restrict__ kv) {
  size_t i = (size_t)blockIdx.x * 256 + threadIdx.x;   // B*KV*C total
  int c = (int)(i % C);
  size_t r = i / C;
  int b = (int)(r / KV), s = (int)(r % KV);
  kv[i] = (s < O) ? memp[((size_t)b * O + s) * C + c]
                  : cur[((size_t)b * T + (s - O)) * C + c];
}

__global__ __launch_bounds__(256)
void rope_kernel(float* __restrict__ buf, int S, int pos_off) {
  size_t i = (size_t)blockIdx.x * 256 + threadIdx.x;   // B*S*H*(HD/2) total
  int j = (int)(i % (HD / 2));
  size_t r = i / (HD / 2);
  int h = (int)(r % H);
  size_t bs = r / H;                 // b*S + s
  int s = (int)(bs % S);
  float fr = expf(-((float)(2 * j) / (float)HD) * 9.210340371976184f);  // ln(1e4)
  float ang = (float)(pos_off + s) * fr;
  float ca = cosf(ang), sa = sinf(ang);
  float* p = buf + bs * (size_t)C + h * HD + 2 * j;
  float x0 = p[0], x1 = p[1];
  p[0] = x0 * ca - x1 * sa;
  p[1] = x0 * sa + x1 * ca;
}

__global__ __launch_bounds__(256)
void newmem_kernel(const float* __restrict__ curn, float* __restrict__ out) {
  size_t i = (size_t)blockIdx.x * 256 + threadIdx.x;   // B*O*C total
  int c = (int)(i % C);
  size_t r = i / C;
  int b = (int)(r / O), j = (int)(r % O);
  out[i] = curn[((size_t)b * T + (T - O + j)) * C + c];
}

__global__ __launch_bounds__(256)
void mask_kernel(const float* __restrict__ curn, const int* __restrict__ tok,
                 float* __restrict__ out) {
  size_t i = (size_t)blockIdx.x * 256 + threadIdx.x;   // B*T*C total
  size_t bt = i / C;
  out[i] = (tok[bt] != 0) ? curn[i] : 0.f;
}

__global__ __launch_bounds__(256)
void ssum_kernel(const float* __restrict__ masked, const int* __restrict__ tok,
                 const float* __restrict__ ssum, const float* __restrict__ lamg,
                 float* __restrict__ out) {
  int i = blockIdx.x * 256 + threadIdx.x;              // B*C total
  int c = i % C, b = i / C;
  float s = 0.f, cnt = 0.f;
  for (int t = 0; t < T; ++t) {
    s   += masked[((size_t)b * T + t) * C + c];
    cnt += (tok[b * T + t] != 0) ? 1.f : 0.f;
  }
  float lam = 1.f / (1.f + expf(-lamg[c]));
  out[i] = ssum[i] * lam + (s / (cnt + 1e-6f)) * (1.f - lam);
}

// ---------------------------------------------------------------------------
extern "C" void kernel_launch(void* const* d_in, const int* in_sizes, int n_in,
                              void* d_out, int out_size, void* d_ws, size_t ws_size,
                              hipStream_t stream) {
  (void)in_sizes; (void)n_in; (void)out_size; (void)ws_size;
  const int*   tok  = (const int*)d_in[0];
  const float* mem  = (const float*)d_in[1];
  const float* ssum = (const float*)d_in[2];
  const float* emb  = (const float*)d_in[3];
  const float* maw  = (const float*)d_in[4];
  const float* mab  = (const float*)d_in[5];
  const float* mns  = (const float*)d_in[6];
  const float* mnb  = (const float*)d_in[7];
  const float* ag   = (const float*)d_in[8];
  const float* lg   = (const float*)d_in[9];
  const float* spw  = (const float*)d_in[10];
  const float* ln1s = (const float*)d_in[11];
  const float* ln1b = (const float*)d_in[12];
  const float* lnks = (const float*)d_in[13];
  const float* lnkb = (const float*)d_in[14];
  const float* qw   = (const float*)d_in[15];
  const float* qbb  = (const float*)d_in[16];
  const float* kw   = (const float*)d_in[17];
  const float* kbb  = (const float*)d_in[18];
  const float* vw   = (const float*)d_in[19];
  const float* vbb  = (const float*)d_in[20];
  const float* ow   = (const float*)d_in[21];
  const float* obb  = (const float*)d_in[22];
  const float* ln2s = (const float*)d_in[23];
  const float* ln2b = (const float*)d_in[24];
  const float* w1   = (const float*)d_in[25];
  const float* b1   = (const float*)d_in[26];
  const float* w2   = (const float*)d_in[27];
  const float* b2   = (const float*)d_in[28];
  const float* fls  = (const float*)d_in[29];
  const float* flb  = (const float*)d_in[30];
  const float* outw = (const float*)d_in[31];
  const float* outb = (const float*)d_in[32];

  float* ws = (float*)d_ws;
  size_t off = 0;
  float* cur  = ws + off; off += (size_t)B * T * C;
  float* nx   = ws + off; off += (size_t)B * T * C;
  float* kv   = ws + off; off += (size_t)B * KV * C;
  float* nkv  = ws + off; off += (size_t)B * KV * C;
  float* qb   = ws + off; off += (size_t)B * T * C;
  float* kb   = ws + off; off += (size_t)B * KV * C;
  float* vb   = ws + off; off += (size_t)B * KV * C;
  float* ab   = ws + off; off += (size_t)B * T * C;
  float* hb   = ws + off; off += (size_t)B * T * 4 * C;
  float* memp = ws + off; off += (size_t)B * O * C;

  dim3 blk(256);
  auto gemm = [&](const float* A, const float* Bw, const float* bias,
                  const float* res, float* Outp, int M, int N, int K, int act) {
    dim3 grid(N / 128, (M + 63) / 64);
    gemm_wmma<<<grid, blk, 0, stream>>>(A, Bw, bias, res, Outp, M, N, K, act);
  };

  // x = embed[tokens] + (ssum @ ssum_proj) * sigmoid(alpha)
  gemm(ssum, spw, nullptr, nullptr, hb, B, C, C, 0);
  embed_kernel<<<(B * T * C) / 256, blk, 0, stream>>>(tok, emb, hb, ag, cur);

  // mem_p = LN(mem @ mem_adapter + b)
  gemm(mem, maw, mab, nullptr, hb, B * O, C, C, 0);
  ln_kernel<<<B * O, blk, 0, stream>>>(hb, mns, mnb, memp);

  for (int i = 0; i < L; ++i) {
    ln_kernel<<<B * T, blk, 0, stream>>>(cur, ln1s + i * C, ln1b + i * C, nx);
    concat_kernel<<<(B * KV * C) / 256, blk, 0, stream>>>(memp, cur, kv);
    ln_kernel<<<B * KV, blk, 0, stream>>>(kv, lnks + i * C, lnkb + i * C, nkv);

    gemm(nx,  qw + (size_t)i * C * C, qbb + i * C, nullptr, qb, B * T,  C, C, 0);
    gemm(nkv, kw + (size_t)i * C * C, kbb + i * C, nullptr, kb, B * KV, C, C, 0);
    gemm(nkv, vw + (size_t)i * C * C, vbb + i * C, nullptr, vb, B * KV, C, C, 0);

    rope_kernel<<<(B * T * H * (HD / 2)) / 256, blk, 0, stream>>>(qb, T, O + 10);
    rope_kernel<<<(B * KV * H * (HD / 2)) / 256, blk, 0, stream>>>(kb, KV, 10);

    attn_wmma<<<dim3(T / 16 / 8, H, B), blk, 0, stream>>>(qb, kb, vb, tok, ab);

    // cur = cur + attn @ o_w + o_b
    gemm(ab, ow + (size_t)i * C * C, obb + i * C, cur, cur, B * T, C, C, 0);

    ln_kernel<<<B * T, blk, 0, stream>>>(cur, ln2s + i * C, ln2b + i * C, nx);
    gemm(nx, w1 + (size_t)i * C * 4 * C, b1 + (size_t)i * 4 * C, nullptr, hb,
         B * T, 4 * C, C, 1);
    gemm(hb, w2 + (size_t)i * 4 * C * C, b2 + i * C, cur, cur, B * T, C, 4 * C, 0);
  }

  // final LN + outputs
  ln_kernel<<<B * T, blk, 0, stream>>>(cur, fls, flb, nx);
  float* out_f = (float*)d_out;
  newmem_kernel<<<(B * O * C) / 256, blk, 0, stream>>>(nx, out_f);
  mask_kernel<<<(B * T * C) / 256, blk, 0, stream>>>(nx, tok, qb);
  ssum_kernel<<<(B * C) / 256, blk, 0, stream>>>(qb, tok, ssum, lg,
                                                 out_f + (size_t)B * O * C);
  gemm(qb, outw, outb, nullptr, out_f + (size_t)B * O * C + B * C, B * T, V, C, 0);
}
